// QuantumLLPModel_11905649345100
// MI455X (gfx1250) — matrix operand order repaired and checked by reading, not compiled, not run on hardware
//
#include <hip/hip_runtime.h>

typedef __attribute__((ext_vector_type(16))) _Float16 v16h;
typedef __attribute__((ext_vector_type(8)))  float    v8f;
typedef __attribute__((ext_vector_type(4)))  float    v4f;

#define QLLP_Q 20
#define QLLP_C 64

__global__ __launch_bounds__(256)
void qllp_wmma_kernel(const float* __restrict__ x,
                      const float* __restrict__ params,
                      const int* __restrict__ patterns,
                      float* __restrict__ out,
                      int nTiles)
{
  const int lane = threadIdx.x & 31;
  const int half = lane >> 4;        // 0: lanes 0-15, 1: lanes 16-31
  const int l15  = lane & 15;

  // ---- A = pattern matrices (4 tiles of 16 classes x 32 K), once per wave ----
  // A 16x32 f16 layout: slot i -> K = i + 16*half ; row M = l15
  v16h amat[4];
  #pragma unroll
  for (int t = 0; t < 4; ++t) {
    const int cls = t * 16 + l15;
    const int* prow = patterns + cls * QLLP_Q;
    #pragma unroll
    for (int i = 0; i < 16; ++i) {
      const int k = i + half * 16;
      const float bit = (k < QLLP_Q) ? (float)prow[k] : 0.0f;
      amat[t][i] = (_Float16)bit;
    }
  }

  // params (20 floats) cached in registers; wave-uniform scalar loads
  float prm[QLLP_Q];
  #pragma unroll
  for (int q = 0; q < QLLP_Q; ++q) prm[q] = params[q];

  const int waveId = (blockIdx.x * (blockDim.x >> 5)) + (threadIdx.x >> 5);
  const int nWaves = gridDim.x * (blockDim.x >> 5);
  // next grid-stride tile is a *constant* element offset from this tile's row
  const long strideElems = (long)nWaves * 16 * QLLP_Q;

  for (int tile = waveId; tile < nTiles; tile += nWaves) {
    const int rowBase = tile * 16;                 // 16 samples per tile (N dim)
    const long row = rowBase + l15;
    const float* xr = x + row * (long)QLLP_Q;

    // overlap next tile's HBM latency with this tile's transcendental work;
    // address = current lane address + uniform constant -> near-zero VALU cost
    if (tile + nWaves < nTiles)
      __builtin_prefetch(xr + half * 16 + strideElems, 0, 0); // global_prefetch_b8

    // ---- B = per-sample log2-ratios (32 K x 16 samples) ----
    // B 32x16 f16 layout: slot i -> K = i + 16*half ; column N = l15
    // half0 lanes: k = 0..15 (4x b128 loads); half1 lanes: k = 16..19 (1x b128)
    v16h bmat;
    #pragma unroll
    for (int i = 0; i < 16; ++i) {
      const int k = i + half * 16;
      float dv = 0.0f;
      if (k < QLLP_Q) {
        const float ang = 0.5f * (3.14159265358979323846f * xr[k] + prm[k]);
        const float s = __sinf(ang);
        const float c = __cosf(ang);
        dv = __log2f(s * s) - __log2f(c * c);     // log2(p1/p0)
        dv = fminf(fmaxf(dv, -100.0f), 100.0f);   // single v_med3_num_f32
      }
      bmat[i] = (_Float16)dv;
    }

    // ---- scores s2[c,b] = sum_q pattern[c,q] * d2[b,q] (base-2 log space) ----
    v8f acc[4];
    #pragma unroll
    for (int t = 0; t < 4; ++t) {
      v8f z = {};
      acc[t] = __builtin_amdgcn_wmma_f32_16x16x32_f16(false, amat[t], false, bmat,
                                                      (short)0, z, false, false);
    }

    // ---- softmax over 64 classes per sample ----
    // D layout: acc[t][m] = score(class t*16 + m + 8*half, sample rowBase+l15)
    // lane l holds 32 classes of sample l15, lane l^16 the other 32.
    // class 0 has all-zero pattern -> score 0 -> e=1, so the sum is >= 1.
    float ebuf[4][8];
    float partial = 0.0f;
    #pragma unroll
    for (int t = 0; t < 4; ++t) {
      #pragma unroll
      for (int m = 0; m < 8; ++m) {
        // direct v_med3_f32 (no canonicalize); scores lie in [-2000, 2000];
        // clamp keeps exp2 finite: 64 * 2^120 < f32 max
        const float sc = __builtin_amdgcn_fmed3f(acc[t][m], -1000.0f, 120.0f);
        const float ee = __builtin_amdgcn_exp2f(sc);   // native v_exp_f32 (exp2)
        ebuf[t][m] = ee;
        partial += ee;
      }
    }
    const float total = partial + __shfl_xor(partial, 16, 32); // single DS op
    const float inv = __builtin_amdgcn_rcpf(total);

    // ---- contiguous b128 stores: 8 consecutive classes per acc ----
    float* orow = out + (long)(rowBase + l15) * QLLP_C + half * 8;
    #pragma unroll
    for (int t = 0; t < 4; ++t) {
      v4f s0 = { ebuf[t][0] * inv, ebuf[t][1] * inv,
                 ebuf[t][2] * inv, ebuf[t][3] * inv };
      v4f s1 = { ebuf[t][4] * inv, ebuf[t][5] * inv,
                 ebuf[t][6] * inv, ebuf[t][7] * inv };
      *(v4f*)(orow + t * 16)     = s0;   // 16B-aligned: offsets are mult. of 32B
      *(v4f*)(orow + t * 16 + 4) = s1;
    }
  }
}

extern "C" void kernel_launch(void* const* d_in, const int* in_sizes, int n_in,
                              void* d_out, int out_size, void* d_ws, size_t ws_size,
                              hipStream_t stream) {
  const float* x        = (const float*)d_in[0];
  const float* params   = (const float*)d_in[1];
  const int*   patterns = (const int*)d_in[2];
  float* out = (float*)d_out;

  const int B = in_sizes[0] / QLLP_Q;   // 65536
  const int nTiles = B / 16;            // 4096 sample-tiles of 16

  const int threads = 256;              // 8 waves / block
  const int wavesPerBlock = threads / 32;
  int targetWaves = (nTiles + 1) / 2;   // ~2 tiles per wave amortize A-build
  int blocks = (targetWaves + wavesPerBlock - 1) / wavesPerBlock;
  if (blocks < 1) blocks = 1;

  qllp_wmma_kernel<<<blocks, threads, 0, stream>>>(x, params, patterns, out, nTiles);
}
